// Graph_MFN_53575422050405
// MI455X (gfx1250) — matrix operand order repaired and checked by reading, not compiled, not run on hardware
//
#include <hip/hip_runtime.h>

// ---------------------------------------------------------------------------
// Graph-MFN on MI455X (gfx1250).
// Every linear layer -> v_wmma_f32_16x16x32_f16 (f16 in, f32 accumulate).
// B=128 rows == 8 M-tiles == one 256-thread block (8 wave32s); N tiled by grid.
// Weights are packed ONCE per call into the exact WMMA B-fragment layout
// (f16, zero-padded), so the K-loop has no bounds checks and no scalar loads:
//   A: two unguarded 32B v8f loads/lane, B: one 32B v16h load/lane.
// Packed weights (~8.8MB) + activations stay resident in the 192MB L2, so the
// 200-step recurrence never touches HBM after step 0.
// ---------------------------------------------------------------------------

typedef __attribute__((ext_vector_type(16))) _Float16 v16h;
typedef __attribute__((ext_vector_type(8)))  float    v8f;

#define BATCH 128
#define HDIM  256
#define TSTEPS 200

// ---------------------------------------------------------------------------
// Weight packing: W is [N,K] row-major f32.  Packed layout, per 16(N)x32(K)
// tile (nt,kt): element index ((nt*KT + kt)*32 + lane)*16 + e holds
// f16( W[nt*16 + (lane&15)][kt*32 + 16*(e>>3) + 8*(lane>>4) + (e&7)] ), zero
// outside [N,K].  B-matrix layout of V_WMMA_F32_16X16X32_F16 (lane->N, elem->K).
// ---------------------------------------------------------------------------
__global__ void pack_w(const float* __restrict__ W, _Float16* __restrict__ out,
                       int N, int K, int KT, int total) {
  const int idx = blockIdx.x * blockDim.x + threadIdx.x;
  if (idx >= total) return;
  const int e    = idx & 15;
  const int lane = (idx >> 4) & 31;
  const int t    = idx >> 9;       // nt*KT + kt
  const int kt   = t % KT;
  const int nt   = t / KT;
  const int n = nt * 16 + (lane & 15);
  const int k = kt * 32 + ((e >> 3) << 4) + ((lane >> 4) << 3) + (e & 7);
  const float v = (n < N && k < K) ? W[(size_t)n * K + k] : 0.f;
  out[idx] = (_Float16)v;
}

struct GemmArgs {
  const float*    A;     int arow;   // f32 activations, row stride (32B-aligned)
  const _Float16* B;                 // packed weights
  int KTfull;                        // k-tiles per packed row of tiles
  int ktBase;                        // first k-tile (for concat slices)
  int KT;                            // k-tiles to process (K padded to 32)
  const float* bias;                 // len N or null
  const float* bias2;                // or null (bih+bhh fold)
  const float* Cin;  int ldc;        // additive input or null
  const float* scale;int scstr;      // per-A-row scalar or null
  float*       Cout; int ldo;
  int M, N;
  int act;                           // 0 none, 1 relu, 2 sigmoid, 3 tanh
};

__device__ __forceinline__ float actf(float x, int act) {
  switch (act) {
    case 1:  return x > 0.f ? x : 0.f;
    case 2:  return 1.f / (1.f + __expf(-x));
    case 3:  return tanhf(x);
    default: return x;
  }
}

// A 16x32 f16 fragment: lane holds row M = m0+(lane&15); element pairs cover
// K chunks [half*8, half*8+8) and [16+half*8, 16+half*8+8) -> two v8f loads.
__device__ __forceinline__ v16h load_a(const float* __restrict__ Arow,
                                       float s, int kb) {
  const v8f x0 = *(const v8f*)(Arow + kb);
  const v8f x1 = *(const v8f*)(Arow + kb + 16);
  v16h a;
#pragma unroll
  for (int i = 0; i < 8; ++i) {
    a[i]     = (_Float16)(x0[i] * s);
    a[i + 8] = (_Float16)(x1[i] * s);
  }
  return a;
}

__global__ __launch_bounds__(256) void gemm_wmma(GemmArgs g) {
  const int wave = threadIdx.x >> 5;
  const int lane = threadIdx.x & 31;
  const int half = lane >> 4;
  const int l16  = lane & 15;
  const int m0 = wave * 16;
  const int n0 = blockIdx.x * 16;
  if (m0 >= g.M) return;  // uniform per wave: EXEC stays all-ones for WMMA

  const int am = m0 + l16;
  const float* Arow = g.A + (size_t)am * g.arow;
  float s = 1.0f;
  if (g.scale) s = g.scale[(size_t)am * g.scstr];

  // lane's slice of the packed B tiles for this n-tile
  const v16h* Bp = (const v16h*)g.B +
                   ((size_t)(n0 >> 4) * g.KTfull + g.ktBase) * 32 + lane;

  v8f acc0 = {}, acc1 = {};
  const int kh = half << 3;
  int kt = 0;
  for (; kt + 1 < g.KT; kt += 2) {
    __builtin_prefetch((const void*)(Bp + (kt + 2) * 32), 0, 0);
    const v16h a0 = load_a(Arow, s, (kt << 5) + kh);
    const v16h b0 = Bp[kt * 32];
    acc0 = __builtin_amdgcn_wmma_f32_16x16x32_f16(
        false, a0, false, b0, (short)0, acc0, false, false);
    const v16h a1 = load_a(Arow, s, ((kt + 1) << 5) + kh);
    const v16h b1 = Bp[(kt + 1) * 32];
    acc1 = __builtin_amdgcn_wmma_f32_16x16x32_f16(
        false, a1, false, b1, (short)0, acc1, false, false);
  }
  if (kt < g.KT) {
    const v16h a0 = load_a(Arow, s, (kt << 5) + kh);
    const v16h b0 = Bp[kt * 32];
    acc0 = __builtin_amdgcn_wmma_f32_16x16x32_f16(
        false, a0, false, b0, (short)0, acc0, false, false);
  }
  const v8f acc = acc0 + acc1;

  // C 16x16 f32: lane -> N = n0+(lane&15); VGPR j -> M = m0 + j + 8*half
  const int n = n0 + l16;
  if (n < g.N) {
    float badd = 0.f;
    if (g.bias)  badd += g.bias[n];
    if (g.bias2) badd += g.bias2[n];
#pragma unroll
    for (int j = 0; j < 8; ++j) {
      const int m = m0 + j + (half << 3);
      float v = acc[j] + badd;
      if (g.Cin) v += g.Cin[(size_t)m * g.ldc + n];
      g.Cout[(size_t)m * g.ldo + n] = actf(v, g.act);
    }
  }
}

// LSTM elementwise: gates [B,4H] (i|f|g|o), c_prev -> h_new, c_new
__global__ void lstm_elem(const float* __restrict__ g,
                          const float* __restrict__ cprev,
                          float* __restrict__ hnew,
                          float* __restrict__ cnew) {
  const int idx = blockIdx.x * blockDim.x + threadIdx.x;
  if (idx >= BATCH * HDIM) return;
  const int b = idx / HDIM, j = idx % HDIM;
  const float* gr = g + (size_t)b * 4 * HDIM;
  const float gi = gr[j];
  const float gf = gr[HDIM + j];
  const float gg = gr[2 * HDIM + j];
  const float go = gr[3 * HDIM + j];
  const float si = 1.f / (1.f + __expf(-gi));
  const float sf = 1.f / (1.f + __expf(-gf));
  const float so = 1.f / (1.f + __expf(-go));
  const float c2 = sf * cprev[idx] + si * tanhf(gg);
  cnew[idx] = c2;
  hnew[idx] = so * tanhf(c2);
}

__global__ void mem_update(const float* __restrict__ g1,
                           const float* __restrict__ g2,
                           const float* __restrict__ cHat,
                           const float* __restrict__ memp,
                           float* __restrict__ memn) {
  const int idx = blockIdx.x * blockDim.x + threadIdx.x;
  if (idx >= BATCH * HDIM) return;
  memn[idx] = g1[idx] * memp[idx] + g2[idx] * cHat[idx];
}

__global__ void zero_fill(float* p, int n) {
  const int idx = blockIdx.x * blockDim.x + threadIdx.x;
  if (idx < n) p[idx] = 0.f;
}

// ---------------------------------------------------------------------------
// Host-side orchestration
// ---------------------------------------------------------------------------

struct PW { const _Float16* p; int KT; };

static inline void gemm(hipStream_t st, const float* A, int arow,
                        PW w, int ktBase, int KT,
                        const float* bias, const float* bias2,
                        const float* Cin, int ldc,
                        const float* scale, int scstr,
                        float* Cout, int ldo, int N, int act) {
  GemmArgs g{A, arow, w.p, w.KT, ktBase, KT, bias, bias2, Cin, ldc,
             scale, scstr, Cout, ldo, BATCH, N, act};
  dim3 grid((N + 15) / 16);
  gemm_wmma<<<grid, 256, 0, st>>>(g);
}

struct Cat { const float* A; const float* scale; };

// Linear over a concat of K=256 pieces (slices = whole k-tile ranges of the
// packed weight), optional per-row scaling; bias+activation on last piece.
static inline void lin_cat(hipStream_t st, const Cat* c, int n, PW w,
                           const float* bias, float* out, int ldo,
                           int N, int act) {
  for (int i = 0; i < n; ++i) {
    gemm(st, c[i].A, HDIM, w, i * 8, 8,
         (i == n - 1) ? bias : nullptr, nullptr,
         (i == 0) ? nullptr : out, ldo,
         c[i].scale, 19,
         out, ldo, N, (i == n - 1) ? act : 0);
  }
}

extern "C" void kernel_launch(void* const* d_in, const int* in_sizes, int n_in,
                              void* d_out, int out_size, void* d_ws, size_t ws_size,
                              hipStream_t stream) {
  (void)in_sizes; (void)n_in; (void)out_size; (void)ws_size;
  const float* texts  = (const float*)d_in[0];  // [128,200,768]
  const float* audios = (const float*)d_in[1];  // [128,200,512]
  const float* videos = (const float*)d_in[2];  // [128,200,256]

  int ix = 3;
#define NEXT ((const float*)d_in[ix++])
  const float *Wih_l = NEXT, *Whh_l = NEXT, *bih_l = NEXT, *bhh_l = NEXT;
  const float *Wih_a = NEXT, *Whh_a = NEXT, *bih_a = NEXT, *bhh_a = NEXT;
  const float *Wih_v = NEXT, *Whh_v = NEXT, *bih_v = NEXT, *bhh_v = NEXT;
  const float *ltrW = NEXT, *ltrb = NEXT;
  const float *atrW = NEXT, *atrb = NEXT;
  const float *vtrW = NEXT, *vtrb = NEXT;
  const float *n01_1W = NEXT, *n01_1b = NEXT, *n01_2W = NEXT, *n01_2b = NEXT;
  const float *n02_1W = NEXT, *n02_1b = NEXT, *n02_2W = NEXT, *n02_2b = NEXT;
  const float *n12_1W = NEXT, *n12_1b = NEXT, *n12_2W = NEXT, *n12_2b = NEXT;
  const float *n012_1W = NEXT, *n012_1b = NEXT, *n012_2W = NEXT, *n012_2b = NEXT;
  const float *t1W = NEXT, *t1b = NEXT, *t2W = NEXT, *t2b = NEXT;
  const float *eff1W = NEXT, *eff1b = NEXT;
  const float *eff2W = NEXT, *eff2b = NEXT;
  const float *eff3W = NEXT, *eff3b = NEXT;
  const float *att1W = NEXT, *att1b = NEXT, *att2W = NEXT, *att2b = NEXT;
  const float *g1f1W = NEXT, *g1f1b = NEXT, *g1f2W = NEXT, *g1f2b = NEXT;
  const float *g2f1W = NEXT, *g2f1b = NEXT, *g2f2W = NEXT, *g2f2b = NEXT;
  const float *of1W = NEXT, *of1b = NEXT, *of2W = NEXT, *of2b = NEXT;
  const float *fo1W = NEXT, *fo1b = NEXT, *fo2W = NEXT, *fo2b = NEXT;
#undef NEXT

  // ---- workspace: f32 activations (32B-aligned), then packed f16 weights
  float* ws = (float*)d_ws;
  size_t off = 0;
  auto allocf = [&](size_t n) {
    float* p = ws + off; off += (n + 7) & ~(size_t)7; return p;
  };
  const size_t S = (size_t)BATCH * HDIM;  // 32768

  float* hb[3][2]; float* cb[3][2]; float* memb[2];
  for (int m = 0; m < 3; ++m) { hb[m][0] = allocf(S); hb[m][1] = allocf(S); }
  for (int m = 0; m < 3; ++m) { cb[m][0] = allocf(S); cb[m][1] = allocf(S); }
  memb[0] = allocf(S); memb[1] = allocf(S);

  float* gbuf[3];
  for (int m = 0; m < 3; ++m) gbuf[m] = allocf((size_t)BATCH * 4 * HDIM);
  float* ls   = allocf(S);
  float* as_  = allocf(S);
  float* vs   = allocf(S);
  float* effh   = allocf((size_t)BATCH * 128);  // 100 -> ld 128, zero-padded
  float* effmid = allocf(S);
  float* effb   = allocf((size_t)BATCH * 19 + 8);
  float* p100   = allocf((size_t)BATCH * 128);  // 100 -> ld 128, zero-padded
  float* o01  = allocf(S);
  float* o02  = allocf(S);
  float* o12  = allocf(S);
  float* o012 = allocf(S);
  float* attended = allocf(S);
  float* atth = allocf(S);
  float* cHat = allocf(S);
  float* gh   = allocf(S);
  float* g1b  = allocf(S);
  float* g2b  = allocf(S);
  float* fh   = allocf(S);
  const size_t zeroN = off;  // zero all f32 state/activations (incl. K pads)

  _Float16* hbase = (_Float16*)(ws + off);
  size_t hoff = 0;
  auto alloch = [&](size_t n) {
    _Float16* p = hbase + hoff; hoff += (n + 15) & ~(size_t)15; return p;
  };

  // deterministic init: recurrent state + padded activation columns
  zero_fill<<<(int)((zeroN + 255) / 256), 256, 0, stream>>>(ws, (int)zeroN);

  // ---- pack all weights (f16 WMMA-B layout, zero-padded), once per call
  auto packW = [&](const float* W, int N, int K) -> PW {
    const int NT = (N + 15) >> 4, KT = (K + 31) >> 5;
    const size_t total = (size_t)NT * KT * 512;
    _Float16* p = alloch(total);
    pack_w<<<(int)((total + 255) / 256), 256, 0, stream>>>(W, p, N, K, KT,
                                                           (int)total);
    return PW{p, KT};
  };

  PW pWih[3] = {packW(Wih_l, 1024, 768), packW(Wih_a, 1024, 512),
                packW(Wih_v, 1024, 256)};
  PW pWhh[3] = {packW(Whh_l, 1024, 256), packW(Whh_a, 1024, 256),
                packW(Whh_v, 1024, 256)};
  PW pltr = packW(ltrW, 256, 512);
  PW patr = packW(atrW, 256, 512);
  PW pvtr = packW(vtrW, 256, 512);
  PW pn01a = packW(n01_1W, 100, 512),  pn01b = packW(n01_2W, 256, 100);
  PW pn02a = packW(n02_1W, 100, 512),  pn02b = packW(n02_2W, 256, 100);
  PW pn12a = packW(n12_1W, 100, 512),  pn12b = packW(n12_2W, 256, 100);
  PW pn3a  = packW(n012_1W, 100, 1536), pn3b = packW(n012_2W, 256, 100);
  PW pt1   = packW(t1W, 100, 1792),     pt2  = packW(t2W, 256, 100);
  PW pe1 = packW(eff1W, 100, 768);
  PW pe2 = packW(eff2W, 256, 100);
  PW pe3 = packW(eff3W, 19, 256);
  PW pat1 = packW(att1W, 256, 256), pat2 = packW(att2W, 256, 256);
  PW pg11 = packW(g1f1W, 256, 512), pg12 = packW(g1f2W, 256, 256);
  PW pg21 = packW(g2f1W, 256, 512), pg22 = packW(g2f2W, 256, 256);
  PW pof1 = packW(of1W, 256, 1024), pof2 = packW(of2W, 128, 256);
  PW pfo1 = packW(fo1W, 6, 128),    pfo2 = packW(fo2W, 1, 128);

  const float* Wbih[3] = {bih_l, bih_a, bih_v};
  const float* Wbhh[3] = {bhh_l, bhh_a, bhh_v};
  const float* xin[3]  = {texts, audios, videos};
  const int    xdim[3] = {768, 512, 256};

  const int EW = 128;  // elementwise grid (128*256/256)

  int cur = 0;
  for (int t = 0; t < TSTEPS; ++t) {
    const int nx = cur ^ 1;

    // ---- LSTM gates + cell (3 modalities) -----------------------------
    for (int m = 0; m < 3; ++m) {
      const float* xt = xin[m] + (size_t)t * xdim[m];  // row b stride T*D
      gemm(stream, xt, TSTEPS * xdim[m], pWih[m], 0, xdim[m] / 32,
           Wbih[m], Wbhh[m], nullptr, 0, nullptr, 0,
           gbuf[m], 4 * HDIM, 4 * HDIM, 0);
      gemm(stream, hb[m][cur], HDIM, pWhh[m], 0, 8,
           nullptr, nullptr, gbuf[m], 4 * HDIM, nullptr, 0,
           gbuf[m], 4 * HDIM, 4 * HDIM, 0);
      lstm_elem<<<EW, 256, 0, stream>>>(gbuf[m], cb[m][cur],
                                        hb[m][nx], cb[m][nx]);
    }

    // ---- modality transforms: relu(W [h_prev, h_new] + b) -------------
    { Cat c[2] = {{hb[0][cur], nullptr}, {hb[0][nx], nullptr}};
      lin_cat(stream, c, 2, pltr, ltrb, ls,  HDIM, HDIM, 1); }
    { Cat c[2] = {{hb[1][cur], nullptr}, {hb[1][nx], nullptr}};
      lin_cat(stream, c, 2, patr, atrb, as_, HDIM, HDIM, 1); }
    { Cat c[2] = {{hb[2][cur], nullptr}, {hb[2][nx], nullptr}};
      lin_cat(stream, c, 2, pvtr, vtrb, vs,  HDIM, HDIM, 1); }

    // ---- efficacy chain: 3H -> 100 -> H -> 19 -------------------------
    { Cat c[3] = {{ls, nullptr}, {as_, nullptr}, {vs, nullptr}};
      lin_cat(stream, c, 3, pe1, eff1b, effh, 128, 100, 0); }
    gemm(stream, effh, 128, pe2, 0, 4, eff2b, nullptr, nullptr, 0,
         nullptr, 0, effmid, HDIM, HDIM, 0);
    gemm(stream, effmid, HDIM, pe3, 0, 8, eff3b, nullptr, nullptr, 0,
         nullptr, 0, effb, 19, 19, 0);
    const float* E = effb;  // per-row scalars, stride 19

    // ---- pairwise nets ------------------------------------------------
    { Cat c[2] = {{ls, E + 0}, {as_, E + 1}};
      lin_cat(stream, c, 2, pn01a, n01_1b, p100, 128, 100, 0); }
    gemm(stream, p100, 128, pn01b, 0, 4, n01_2b, nullptr, nullptr, 0,
         nullptr, 0, o01, HDIM, HDIM, 0);
    { Cat c[2] = {{ls, E + 2}, {vs, E + 3}};
      lin_cat(stream, c, 2, pn02a, n02_1b, p100, 128, 100, 0); }
    gemm(stream, p100, 128, pn02b, 0, 4, n02_2b, nullptr, nullptr, 0,
         nullptr, 0, o02, HDIM, HDIM, 0);
    { Cat c[2] = {{as_, E + 4}, {vs, E + 5}};
      lin_cat(stream, c, 2, pn12a, n12_1b, p100, 128, 100, 0); }
    gemm(stream, p100, 128, pn12b, 0, 4, n12_2b, nullptr, nullptr, 0,
         nullptr, 0, o12, HDIM, HDIM, 0);

    // ---- triple net ---------------------------------------------------
    { Cat c[6] = {{ls, E + 6}, {as_, E + 7}, {vs, E + 8},
                  {o01, E + 9}, {o02, E + 10}, {o12, E + 11}};
      lin_cat(stream, c, 6, pn3a, n012_1b, p100, 128, 100, 0); }
    gemm(stream, p100, 128, pn3b, 0, 4, n012_2b, nullptr, nullptr, 0,
         nullptr, 0, o012, HDIM, HDIM, 0);

    // ---- top net -> attended ------------------------------------------
    { Cat c[7] = {{ls, E + 12}, {as_, E + 13}, {vs, E + 14},
                  {o01, E + 15}, {o02, E + 16}, {o12, E + 17},
                  {o012, E + 18}};
      lin_cat(stream, c, 7, pt1, t1b, p100, 128, 100, 0); }
    gemm(stream, p100, 128, pt2, 0, 4, t2b, nullptr, nullptr, 0,
         nullptr, 0, attended, HDIM, HDIM, 0);

    // ---- cHat = tanh(fc2(relu(fc1(attended)))) ------------------------
    gemm(stream, attended, HDIM, pat1, 0, 8, att1b, nullptr, nullptr, 0,
         nullptr, 0, atth, HDIM, HDIM, 1);
    gemm(stream, atth, HDIM, pat2, 0, 8, att2b, nullptr, nullptr, 0,
         nullptr, 0, cHat, HDIM, HDIM, 3);

    // ---- gates g1,g2 over [attended, mem] -----------------------------
    { Cat c[2] = {{attended, nullptr}, {memb[cur], nullptr}};
      lin_cat(stream, c, 2, pg11, g1f1b, gh, HDIM, HDIM, 1); }
    gemm(stream, gh, HDIM, pg12, 0, 8, g1f2b, nullptr, nullptr, 0,
         nullptr, 0, g1b, HDIM, HDIM, 2);
    { Cat c[2] = {{attended, nullptr}, {memb[cur], nullptr}};
      lin_cat(stream, c, 2, pg21, g2f1b, gh, HDIM, HDIM, 1); }
    gemm(stream, gh, HDIM, pg22, 0, 8, g2f2b, nullptr, nullptr, 0,
         nullptr, 0, g2b, HDIM, HDIM, 2);

    mem_update<<<EW, 256, 0, stream>>>(g1b, g2b, cHat, memb[cur], memb[nx]);
    cur = nx;
  }

  // ---- head: features = fc2(relu(fc1([h_l,h_a,h_v,mem]))) -------------
  { Cat c[4] = {{hb[0][cur], nullptr}, {hb[1][cur], nullptr},
                {hb[2][cur], nullptr}, {memb[cur], nullptr}};
    lin_cat(stream, c, 4, pof1, of1b, fh, HDIM, HDIM, 1); }

  float* out = (float*)d_out;
  float* features = out;                  // [128,128]
  float* emos     = out + 128 * 128;      // [128,6]
  float* vals     = emos + 128 * 6;       // [128,1]
  gemm(stream, fh, HDIM, pof2, 0, 8, of2b, nullptr, nullptr, 0,
       nullptr, 0, features, 128, 128, 0);
  gemm(stream, features, 128, pfo1, 0, 4, fo1b, nullptr, nullptr, 0,
       nullptr, 0, emos, 6, 6, 0);
  gemm(stream, features, 128, pfo2, 0, 4, fo2b, nullptr, nullptr, 0,
       nullptr, 0, vals, 1, 1, 0);
}
